// MultiHeadCrossAttention_352187318939
// MI455X (gfx1250) — matrix operand order repaired and checked by reading, not compiled, not run on hardware
//
#include <hip/hip_runtime.h>

// ---------------- problem constants ----------------
#define B_    16
#define NQ_   512
#define NK_   1024
#define CIN_  128
#define H_    8
#define D_    64
#define HD_   512     // H_*D_

typedef __attribute__((ext_vector_type(16))) _Float16 v16h;
typedef __attribute__((ext_vector_type(8)))  float    v8f;
typedef __attribute__((ext_vector_type(8)))  _Float16 h8;
typedef __attribute__((ext_vector_type(4)))  float    f4;

static __device__ __forceinline__ v8f wmma_f16(v16h a, v16h b, v8f c) {
  // D = A(16x32 f16) * B(32x16 f16) + C(16x16 f32)
  return __builtin_amdgcn_wmma_f32_16x16x32_f16(
      /*neg_a=*/false, a, /*neg_b=*/false, b,
      /*c_mod=*/(short)0, c, /*reuse_a=*/false, /*reuse_b=*/false);
}

static __device__ __forceinline__ v8f vzero8() {
  v8f z;
#pragma unroll
  for (int i = 0; i < 8; ++i) z[i] = 0.0f;
  return z;
}

// Flat (generic) pointer -> LDS byte offset, via addrspace(3) cast.
typedef __attribute__((address_space(3))) const void* lds_cptr_t;
static __device__ __forceinline__ unsigned lds_offset(const void* p) {
  return (unsigned)(unsigned long long)(lds_cptr_t)p;
}

// CDNA5 async copy: 16B global -> LDS, tracked by ASYNCcnt (no VGPR round-trip)
static __device__ __forceinline__ void async_g2l_b128(unsigned lds_off,
                                                      const void* gaddr) {
  asm volatile("global_load_async_to_lds_b128 %0, %1, off"
               :: "v"(lds_off), "v"((unsigned long long)gaddr)
               : "memory");
}
static __device__ __forceinline__ void wait_async0() {
  asm volatile("s_wait_asynccnt 0x0" ::: "memory");
}

// Build a v16h fragment from two contiguous 8-element halves
static __device__ __forceinline__ v16h frag16(const _Float16* lo_p,
                                              const _Float16* hi_p) {
  h8 lo = *(const h8*)lo_p;
  h8 hi = *(const h8*)hi_p;
  v16h f;
#pragma unroll
  for (int i = 0; i < 8; ++i) { f[i] = lo[i]; f[8 + i] = hi[i]; }
  return f;
}

// =====================================================================
// Prep 1: f32 -> f16 elementwise (activations), 4 elems/thread
// =====================================================================
__global__ __launch_bounds__(256) void cvt_x_kernel(
    const float* __restrict__ src, _Float16* __restrict__ dst, int n4) {
  const int i = blockIdx.x * 256 + threadIdx.x;   // index in float4 units
  if (i >= n4) return;
  f4 v = *(const f4*)(src + (size_t)i * 4);
  _Float16* d = dst + (size_t)i * 4;
#pragma unroll
  for (int j = 0; j < 4; ++j) d[j] = (_Float16)v[j];
}

// =====================================================================
// Prep 2: W [128][512] f32 -> Wt [512][128] f16 (transpose + convert)
// =====================================================================
__global__ __launch_bounds__(256) void cvt_w_kernel(
    const float* __restrict__ W, _Float16* __restrict__ Wt) {
  const int idx = blockIdx.x * 256 + threadIdx.x;  // 0..65535
  const int k = idx >> 9;        // row in W  (0..127)
  const int n = idx & 511;       // col in W  (0..511)
  Wt[(size_t)n * CIN_ + k] = (_Float16)W[idx];
}

// =====================================================================
// Projection: y[b,h,n,d] = f16( (x[b,n,:] @ W + bias) * scaleMul )
// x16: [B_*Nrow, 128] f16; Wt: [512][128] f16 (pre-transposed);
// y: [B_, H_, Nrow, 64] f16.  One 16x16 tile/wave: preload all 8
// fragments (one load clause), then chain 4 WMMAs.
// nrowShift: log2(Nrow) -> div/mod become shift/mask.
// =====================================================================
__global__ __launch_bounds__(128) void proj_f16_kernel(
    const _Float16* __restrict__ x16, const _Float16* __restrict__ Wt,
    const float* __restrict__ bias, _Float16* __restrict__ y,
    int nrowShift, float scaleMul) {
  const int Nrow = 1 << nrowShift;
  const int gw   = (int)((blockIdx.x * blockDim.x + threadIdx.x) >> 5);
  const int lane = threadIdx.x & 31;
  const int tM = gw >> 5;          // 32 N-tiles (512/16)
  const int tN = gw & 31;

  const int half = lane >> 4;
  const int l16  = lane & 15;
  const int n0   = tN * 16;
  const _Float16* xr = x16 + (size_t)(tM * 16 + l16) * CIN_;  // A row
  const _Float16* wr = Wt + (size_t)(n0 + l16) * CIN_;        // B col (transposed)

  // Preload all fragments (A and B: k=32c+half*8+e / +16)
  v16h a[4], bw[4];
#pragma unroll
  for (int c = 0; c < 4; ++c) {
    a[c]  = frag16(xr + 32 * c + half * 8, xr + 32 * c + 16 + half * 8);
    bw[c] = frag16(wr + 32 * c + half * 8, wr + 32 * c + 16 + half * 8);
  }
  v8f acc = vzero8();
#pragma unroll
  for (int c = 0; c < 4; ++c) acc = wmma_f16(a[c], bw[c], acc);

  // C layout: VGPR g -> row m = half*8+g, col n = l16
  const int   col    = n0 + l16;
  const float bias_v = bias[col];
  const int   hh = col >> 6;      // head
  const int   dd = col & 63;      // dim within head
#pragma unroll
  for (int g = 0; g < 8; ++g) {
    const int r  = tM * 16 + half * 8 + g;
    const int bi = r >> nrowShift;
    const int nr = r & (Nrow - 1);
    const float v = (acc[g] + bias_v) * scaleMul;
    y[(((size_t)bi * H_ + hh) * Nrow + nr) * D_ + dd] = (_Float16)v;
  }
}

// =====================================================================
// Flash attention: block = 4 waves (128 thr), each wave owns 16 queries.
// 32-key tiles, DOUBLE-BUFFERED in LDS: async K copy + V loads for tile
// t+1 are issued before computing tile t; V-transpose commit + asynccnt
// wait + one barrier per iteration.
// =====================================================================
__global__ __launch_bounds__(128) void attn_kernel(
    const _Float16* __restrict__ qh, const _Float16* __restrict__ kh,
    const _Float16* __restrict__ vh, const float* __restrict__ cmask,
    float* __restrict__ out) {
  __shared__ __attribute__((aligned(16))) _Float16 Kt[2][32 * 64];   // [key][d]
  __shared__ __attribute__((aligned(16))) _Float16 VtT[2][64 * 32];  // [d][key]
  __shared__ __attribute__((aligned(16))) _Float16 Pst[4][16 * 32];  // per-wave

  const int tid  = threadIdx.x;
  const int lane = tid & 31;
  const int wave = tid >> 5;
  const int half = lane >> 4;
  const int l16  = lane & 15;

  const int b = blockIdx.z, hd = blockIdx.y, qblk = blockIdx.x;
  const int q0 = qblk * 64 + wave * 16;
  const size_t headQ = ((size_t)b * H_ + hd) * NQ_;
  const size_t headK = ((size_t)b * H_ + hd) * NK_;

  // per-thread staging slots (tile is 32 keys x 64 dims = 256 vec8 chunks;
  // 128 threads x 2 chunks: keys key0 and key0+16, same 8-dim segment)
  const int key0 = tid >> 3;
  const int ds0  = (tid & 7) << 3;
  const int key1 = key0 + 16;

  // ---- load Q tile into A fragments (kept in registers all loop) ----
  v16h aq[2];
  {
    const _Float16* qr = qh + (headQ + q0 + l16) * D_;
#pragma unroll
    for (int c = 0; c < 2; ++c)
      aq[c] = frag16(qr + 32 * c + half * 8, qr + 32 * c + 16 + half * 8);
  }

  v8f o[4];
#pragma unroll
  for (int j = 0; j < 4; ++j) o[j] = vzero8();
  v8f mrow, lrow;
#pragma unroll
  for (int g = 0; g < 8; ++g) { mrow[g] = -3.0e38f; lrow[g] = 0.0f; }

  // ---- prologue: stage tile 0 into buffer 0 ----
  {
    async_g2l_b128(lds_offset(&Kt[0][key0 * 64 + ds0]),
                   kh + (headK + key0) * D_ + ds0);
    async_g2l_b128(lds_offset(&Kt[0][key1 * 64 + ds0]),
                   kh + (headK + key1) * D_ + ds0);
    h8 va = *(const h8*)(vh + (headK + key0) * D_ + ds0);
    h8 vb = *(const h8*)(vh + (headK + key1) * D_ + ds0);
#pragma unroll
    for (int i = 0; i < 8; ++i) {
      VtT[0][(ds0 + i) * 32 + key0] = va[i];
      VtT[0][(ds0 + i) * 32 + key1] = vb[i];
    }
    wait_async0();
    __syncthreads();
  }

  int cur = 0;
  for (int kb = 0; kb < NK_; kb += 32) {
    const bool more = (kb + 32 < NK_);
    const int  nxt  = cur ^ 1;

    // ---- issue next tile's loads (K async -> LDS, V global -> VGPR) ----
    h8 va, vb;
    if (more) {
      const size_t gk = headK + kb + 32;
      async_g2l_b128(lds_offset(&Kt[nxt][key0 * 64 + ds0]),
                     kh + (gk + key0) * D_ + ds0);
      async_g2l_b128(lds_offset(&Kt[nxt][key1 * 64 + ds0]),
                     kh + (gk + key1) * D_ + ds0);
      va = *(const h8*)(vh + (gk + key0) * D_ + ds0);
      vb = *(const h8*)(vh + (gk + key1) * D_ + ds0);
    }

    // additive mask for this lane's two key columns
    const float madd0 = -100000.0f * (1.0f - cmask[(size_t)b * NK_ + kb + l16]);
    const float madd1 = -100000.0f * (1.0f - cmask[(size_t)b * NK_ + kb + 16 + l16]);

    // ---- scores: S = Q * K^T (two 16x16 N-tiles, K-dim 64 = 2 chunks) ----
    const _Float16* K0 = &Kt[cur][0];
    v8f s0 = vzero8(), s1 = vzero8();
#pragma unroll
    for (int c = 0; c < 2; ++c) {
      const _Float16* kp0 = K0 + l16 * 64 + 32 * c + half * 8;
      const _Float16* kp1 = K0 + (16 + l16) * 64 + 32 * c + half * 8;
      v16h b0 = frag16(kp0, kp0 + 16);
      v16h b1 = frag16(kp1, kp1 + 16);
      s0 = wmma_f16(aq[c], b0, s0);
      s1 = wmma_f16(aq[c], b1, s1);
    }
#pragma unroll
    for (int g = 0; g < 8; ++g) { s0[g] += madd0; s1[g] += madd1; }

    // ---- online softmax (rows live across 16-lane half-groups) ----
    v8f pm;
#pragma unroll
    for (int g = 0; g < 8; ++g) pm[g] = fmaxf(s0[g], s1[g]);
#pragma unroll
    for (int msk = 1; msk < 16; msk <<= 1)
#pragma unroll
      for (int g = 0; g < 8; ++g) pm[g] = fmaxf(pm[g], __shfl_xor(pm[g], msk, 32));

    v8f mnew, alpha, p0, p1, rs;
#pragma unroll
    for (int g = 0; g < 8; ++g) {
      mnew[g]  = fmaxf(mrow[g], pm[g]);
      alpha[g] = __expf(mrow[g] - mnew[g]);
      p0[g] = __expf(s0[g] - mnew[g]);
      p1[g] = __expf(s1[g] - mnew[g]);
      rs[g] = p0[g] + p1[g];
    }
#pragma unroll
    for (int msk = 1; msk < 16; msk <<= 1)
#pragma unroll
      for (int g = 0; g < 8; ++g) rs[g] += __shfl_xor(rs[g], msk, 32);
#pragma unroll
    for (int g = 0; g < 8; ++g) {
      lrow[g] = lrow[g] * alpha[g] + rs[g];
      mrow[g] = mnew[g];
    }
#pragma unroll
    for (int j = 0; j < 4; ++j)
#pragma unroll
      for (int g = 0; g < 8; ++g) o[j][g] *= alpha[g];

    // ---- P (C layout) -> LDS -> A layout (per-wave staging, in-order DS) ----
    _Float16* P = &Pst[wave][0];
#pragma unroll
    for (int g = 0; g < 8; ++g) {
      P[(half * 8 + g) * 32 + l16]      = (_Float16)p0[g];
      P[(half * 8 + g) * 32 + 16 + l16] = (_Float16)p1[g];
    }
    const _Float16* pr = &P[l16 * 32];
    v16h ap = frag16(pr + half * 8, pr + 16 + half * 8);

    // ---- O += P * V  (4 output N-tiles of 16 dims, K-dim = 32 keys) ----
    const _Float16* V0 = &VtT[cur][0];
#pragma unroll
    for (int j = 0; j < 4; ++j) {
      const _Float16* vp = V0 + (j * 16 + l16) * 32 + half * 8;
      v16h bvf = frag16(vp, vp + 16);
      o[j] = wmma_f16(ap, bvf, o[j]);
    }

    // ---- commit next tile: V transpose stores + async fence ----
    if (more) {
#pragma unroll
      for (int i = 0; i < 8; ++i) {
        VtT[nxt][(ds0 + i) * 32 + key0] = va[i];
        VtT[nxt][(ds0 + i) * 32 + key1] = vb[i];
      }
      wait_async0();
    }
    __syncthreads();
    cur = nxt;
  }

  // ---- epilogue: normalize and store [B, NQ, H*D] ----
#pragma unroll
  for (int j = 0; j < 4; ++j) {
#pragma unroll
    for (int g = 0; g < 8; ++g) {
      const int qrow = q0 + half * 8 + g;
      const int dd   = j * 16 + l16;
      out[(((size_t)b * NQ_ + qrow) * H_ + hd) * D_ + dd] = o[j][g] / lrow[g];
    }
  }
}

// =====================================================================
extern "C" void kernel_launch(void* const* d_in, const int* in_sizes, int n_in,
                              void* d_out, int out_size, void* d_ws, size_t ws_size,
                              hipStream_t stream) {
  (void)in_sizes; (void)n_in; (void)out_size; (void)ws_size;
  const float* query = (const float*)d_in[0];
  const float* keyi  = (const float*)d_in[1];
  const float* cmask = (const float*)d_in[2];
  const float* Wq    = (const float*)d_in[3];
  const float* bq    = (const float*)d_in[4];
  const float* Wk    = (const float*)d_in[5];
  const float* bk    = (const float*)d_in[6];
  const float* Wv    = (const float*)d_in[7];
  const float* bv    = (const float*)d_in[8];
  float* out = (float*)d_out;

  // ---- workspace layout (f16) ----
  _Float16* qbf = (_Float16*)d_ws;                          // [B,H,NQ,D]  4Mi
  _Float16* kbf = qbf + (size_t)B_ * H_ * NQ_ * D_;         // [B,H,NK,D]  8Mi
  _Float16* vbf = kbf + (size_t)B_ * H_ * NK_ * D_;         // [B,H,NK,D]  8Mi
  _Float16* xq16 = vbf + (size_t)B_ * H_ * NK_ * D_;        // [B*NQ,128]  1Mi
  _Float16* xk16 = xq16 + (size_t)B_ * NQ_ * CIN_;          // [B*NK,128]  2Mi
  _Float16* wqT  = xk16 + (size_t)B_ * NK_ * CIN_;          // [512][128]
  _Float16* wkT  = wqT + (size_t)HD_ * CIN_;
  _Float16* wvT  = wkT + (size_t)HD_ * CIN_;

  // ---- prep: convert activations and weights to f16 (one pass) ----
  {
    const int nq4 = (B_ * NQ_ * CIN_) / 4;   // 262144
    const int nk4 = (B_ * NK_ * CIN_) / 4;   // 524288
    cvt_x_kernel<<<(nq4 + 255) / 256, 256, 0, stream>>>(query, xq16, nq4);
    cvt_x_kernel<<<(nk4 + 255) / 256, 256, 0, stream>>>(keyi, xk16, nk4);
    cvt_w_kernel<<<(CIN_ * HD_) / 256, 256, 0, stream>>>(Wq, wqT);
    cvt_w_kernel<<<(CIN_ * HD_) / 256, 256, 0, stream>>>(Wk, wkT);
    cvt_w_kernel<<<(CIN_ * HD_) / 256, 256, 0, stream>>>(Wv, wvT);
  }

  const float invScale = 0.125f;  // 1/sqrt(64), folded into Q projection

  // proj Q: Mtiles=512 -> 512*32 tiles / 4 waves = 4096 blocks
  proj_f16_kernel<<<4096, 128, 0, stream>>>(xq16, wqT, bq, qbf, 9, invScale);
  // proj K/V: Mtiles=1024 -> 8192 blocks each
  proj_f16_kernel<<<8192, 128, 0, stream>>>(xk16, wkT, bk, kbf, 10, 1.0f);
  proj_f16_kernel<<<8192, 128, 0, stream>>>(xk16, wvT, bv, vbf, 10, 1.0f);

  // attention: grid (NQ/64, H, B), 128 threads (4 waves x 16 queries)
  attn_kernel<<<dim3(NQ_ / 64, H_, B_), 128, 0, stream>>>(qbf, kbf, vbf, cmask, out);
}
